// CMoE_a_78640851189970
// MI455X (gfx1250) — matrix operand order repaired and checked by reading, not compiled, not run on hardware
//
#include <hip/hip_runtime.h>

// ---- problem constants (fixed by setup_inputs) ----
constexpr int Bb   = 8;
constexpr int Tt   = 2048;
constexpr int Mdim = 1024;   // m
constexpr int Fdim = 2048;   // f
constexpr int Ee   = 8;
constexpr int S    = Bb * Tt;        // 16384 tokens
constexpr int CAP  = 2048;           // ceil(S/E)
constexpr int HASHP = 5099;

typedef __attribute__((ext_vector_type(8)))  __bf16 v8bf;
typedef __attribute__((ext_vector_type(16))) __bf16 v16bf;
typedef __attribute__((ext_vector_type(8)))  float  v8f;

__device__ __forceinline__ __bf16 hi_bf16(float v) { return (__bf16)v; }
__device__ __forceinline__ __bf16 lo_bf16(float v, __bf16 h) {
    return (__bf16)(v - (float)h);
}
__device__ __forceinline__ v16bf cat8(v8bf a, v8bf b) {
    return __builtin_shufflevector(a, b, 0,1,2,3,4,5,6,7,8,9,10,11,12,13,14,15);
}

// ======================================================================
// Kernel 1: hash routing. One workgroup, 256 threads, 64 tokens each.
// ======================================================================
__global__ __launch_bounds__(256)
void route_kernel(const int* __restrict__ tok,
                  int* __restrict__ flat_idx,
                  int* __restrict__ inv) {
    __shared__ int cnts[256][Ee];
    const int tid = threadIdx.x;
    for (int i = 0; i < S / 256; ++i) inv[i * 256 + tid] = -1;

    const int per = S / 256;
    const int base = tid * per;
    int local[Ee];
#pragma unroll
    for (int e = 0; e < Ee; ++e) local[e] = 0;
    for (int i = 0; i < per; ++i) local[(tok[base + i] * HASHP) % Ee]++;
#pragma unroll
    for (int e = 0; e < Ee; ++e) cnts[tid][e] = local[e];
    __syncthreads();
    if (tid < Ee) {
        int run = 0;
        for (int i = 0; i < 256; ++i) { int t = cnts[i][tid]; cnts[i][tid] = run; run += t; }
    }
    __syncthreads();
    int cnt[Ee];
#pragma unroll
    for (int e = 0; e < Ee; ++e) cnt[e] = cnts[tid][e];
    for (int i = 0; i < per; ++i) {
        int s = base + i;
        int e = (tok[s] * HASHP) % Ee;
        int pos = cnt[e]++;
        int fl = (pos < CAP) ? (e * CAP + pos) : -1;
        flat_idx[s] = fl;
        if (fl >= 0) inv[fl] = s;
    }
}

// ======================================================================
// Weight pre-pass: fp32 W[E][K][N]  ->  bf16 hi/lo Wt[E][N][K]
// 32x32 LDS-tiled transpose, coalesced both directions.
// ======================================================================
__global__ __launch_bounds__(256)
void wsplit_kernel(const float* __restrict__ W,
                   __bf16* __restrict__ Th, __bf16* __restrict__ Tl,
                   int K, int N) {
    __shared__ float tile[32][33];
    const int e = blockIdx.z;
    const int kb = blockIdx.y * 32, nb = blockIdx.x * 32;
    const float* We = W + (long)e * K * N;
    const int tx = threadIdx.x & 31, ty = threadIdx.x >> 5;
#pragma unroll
    for (int r = ty; r < 32; r += 8)
        tile[r][tx] = We[(long)(kb + r) * N + nb + tx];
    __syncthreads();
    const long base = (long)e * N * K;
#pragma unroll
    for (int r = ty; r < 32; r += 8) {
        float v = tile[tx][r];                    // element [k=kb+tx][n=nb+r]
        __bf16 h = hi_bf16(v);
        __bf16 l = lo_bf16(v, h);
        long o = base + (long)(nb + r) * K + kb + tx;
        Th[o] = h; Tl[o] = l;
    }
}

// ======================================================================
// bf16x3 split GEMM over dispatched slots (fp32-faithful via hi/lo parts).
// Block tile 128x128, K-step 32, 8 waves, each wave 64x32 (4x2 subtiles),
// 3x v_wmma_f32_16x16x32_bf16 per subtile per K-step.
//   ALOAD 0: A row = token-shift mix of x gathered via inv (fused dispatch)
//   ALOAD 1: A from bf16 hi/lo buffers (h)
//   EPI 0: relu^2 -> bf16 hi/lo   EPI 1: identity -> f32   EPI 2: sigmoid*Cmul -> f32
// ======================================================================
template <int ALOAD, int EPI>
__global__ __launch_bounds__(256)
void gemm_bf16x3(const float*  __restrict__ Af32,
                 const __bf16* __restrict__ Abh,
                 const __bf16* __restrict__ Abl,
                 const int*    __restrict__ inv,
                 const float*  __restrict__ shifts,
                 const float*  __restrict__ maa,
                 const __bf16* __restrict__ Wth,   // [E][N][K]
                 const __bf16* __restrict__ Wtl,
                 const float*  __restrict__ Cmul,
                 float*        __restrict__ Df32,
                 __bf16*       __restrict__ Dh,
                 __bf16*       __restrict__ Dl,
                 int K, int N) {
    __shared__ __bf16 As_hi[128][40];   // 80B row pitch: 16B-aligned rows
    __shared__ __bf16 As_lo[128][40];
    __shared__ __bf16 Bs_hi[128][40];   // rows = n, cols = k
    __shared__ __bf16 Bs_lo[128][40];

    const int bn = blockIdx.x;
    const int bm = blockIdx.y;
    const int e  = bm / (CAP / 128);
    const long wbase = (long)e * N * K;

    const int tid  = threadIdx.x;
    const int lane = tid & 31;
    const int wave = tid >> 5;
    const int wm = wave >> 2;           // 0..1
    const int wn = wave & 3;            // 0..3

    v8f acc[4][2];
#pragma unroll
    for (int mi = 0; mi < 4; ++mi)
#pragma unroll
        for (int ni = 0; ni < 2; ++ni) acc[mi][ni] = (v8f)0.0f;

    // A-tile loader: 2 threads/row, 16 k-elements each
    const int ar   = tid >> 1;
    const int ac   = (tid & 1) * 16;
    const int arow = bm * 128 + ar;

    const float*  xrow = nullptr;
    const float*  prow = nullptr;
    const __bf16* ahrow = nullptr;
    const __bf16* alrow = nullptr;
    if (ALOAD == 0) {
        int src = inv[arow];
        if (src >= 0) {
            xrow = Af32 + (long)src * K;
            prow = ((src % Tt) == 0) ? (shifts + (long)(src / Tt) * K)
                                     : (Af32 + (long)(src - 1) * K);
        }
    } else {
        ahrow = Abh + (long)arow * K;
        alrow = Abl + (long)arow * K;
    }

    // B-tile loader: 2 threads per n-row, 16 k-elements each
    const int bnr = tid >> 1;
    const int bkc = (tid & 1) * 16;
    const __bf16* wh = Wth + wbase + (long)(bn * 128 + bnr) * K;
    const __bf16* wl = Wtl + wbase + (long)(bn * 128 + bnr) * K;

    const int l15  = lane & 15;
    const int half = lane >> 4;

    for (int k0 = 0; k0 < K; k0 += 32) {
        // ---- stage A (fused dispatch-gather + token-shift mix + split) ----
        if (ALOAD == 0) {
            v8bf hh[2], ll[2];
            if (xrow) {
#pragma unroll
                for (int q = 0; q < 4; ++q) {
                    int k = k0 + ac + q * 4;
                    float4 xv = *(const float4*)(xrow + k);
                    float4 pv = *(const float4*)(prow + k);
                    float4 mv = *(const float4*)(maa + k);
                    float f0 = xv.x + (pv.x - xv.x) * mv.x;
                    float f1 = xv.y + (pv.y - xv.y) * mv.y;
                    float f2 = xv.z + (pv.z - xv.z) * mv.z;
                    float f3 = xv.w + (pv.w - xv.w) * mv.w;
                    int g = q >> 1, o = (q & 1) * 4;
                    __bf16 h0 = hi_bf16(f0), l0 = lo_bf16(f0, h0);
                    __bf16 h1 = hi_bf16(f1), l1 = lo_bf16(f1, h1);
                    __bf16 h2 = hi_bf16(f2), l2 = lo_bf16(f2, h2);
                    __bf16 h3 = hi_bf16(f3), l3 = lo_bf16(f3, h3);
                    hh[g][o + 0] = h0; ll[g][o + 0] = l0;
                    hh[g][o + 1] = h1; ll[g][o + 1] = l1;
                    hh[g][o + 2] = h2; ll[g][o + 2] = l2;
                    hh[g][o + 3] = h3; ll[g][o + 3] = l3;
                }
            } else {
                hh[0] = (v8bf)(__bf16)0.0f; hh[1] = hh[0];
                ll[0] = hh[0];              ll[1] = hh[0];
            }
            *(v8bf*)&As_hi[ar][ac + 0] = hh[0];
            *(v8bf*)&As_hi[ar][ac + 8] = hh[1];
            *(v8bf*)&As_lo[ar][ac + 0] = ll[0];
            *(v8bf*)&As_lo[ar][ac + 8] = ll[1];
        } else {
            *(v8bf*)&As_hi[ar][ac + 0] = *(const v8bf*)(ahrow + k0 + ac);
            *(v8bf*)&As_hi[ar][ac + 8] = *(const v8bf*)(ahrow + k0 + ac + 8);
            *(v8bf*)&As_lo[ar][ac + 0] = *(const v8bf*)(alrow + k0 + ac);
            *(v8bf*)&As_lo[ar][ac + 8] = *(const v8bf*)(alrow + k0 + ac + 8);
        }
        // ---- stage B (pre-transposed bf16 weights, contiguous along K) ----
        *(v8bf*)&Bs_hi[bnr][bkc + 0] = *(const v8bf*)(wh + k0 + bkc);
        *(v8bf*)&Bs_hi[bnr][bkc + 8] = *(const v8bf*)(wh + k0 + bkc + 8);
        *(v8bf*)&Bs_lo[bnr][bkc + 0] = *(const v8bf*)(wl + k0 + bkc);
        *(v8bf*)&Bs_lo[bnr][bkc + 8] = *(const v8bf*)(wl + k0 + bkc + 8);
        if (k0 + 32 < K) {                       // next weight K-tile
            __builtin_prefetch(wh + k0 + 32 + bkc, 0, 1);
            __builtin_prefetch(wl + k0 + 32 + bkc, 0, 1);
        }
        __syncthreads();

        // ---- B fragments: per-lane fixed N, K = half*8 + [0,8) and 16 + half*8 + [0,8) ----
        v16bf bh[2], bl[2];
#pragma unroll
        for (int ni = 0; ni < 2; ++ni) {
            int n = wn * 32 + ni * 16 + l15;
            bh[ni] = cat8(*(const v8bf*)&Bs_hi[n][half * 8],
                          *(const v8bf*)&Bs_hi[n][16 + half * 8]);
            bl[ni] = cat8(*(const v8bf*)&Bs_lo[n][half * 8],
                          *(const v8bf*)&Bs_lo[n][16 + half * 8]);
        }
        // ---- A fragments + 3-term split WMMA ----
#pragma unroll
        for (int mi = 0; mi < 4; ++mi) {
            int r = wm * 64 + mi * 16 + l15;
            v16bf ah = cat8(*(const v8bf*)&As_hi[r][half * 8],
                            *(const v8bf*)&As_hi[r][16 + half * 8]);
            v16bf al = cat8(*(const v8bf*)&As_lo[r][half * 8],
                            *(const v8bf*)&As_lo[r][16 + half * 8]);
#pragma unroll
            for (int ni = 0; ni < 2; ++ni) {
                acc[mi][ni] = __builtin_amdgcn_wmma_f32_16x16x32_bf16(
                    false, ah, false, bh[ni], (short)0, acc[mi][ni], false, false);
                acc[mi][ni] = __builtin_amdgcn_wmma_f32_16x16x32_bf16(
                    false, ah, false, bl[ni], (short)0, acc[mi][ni], false, false);
                acc[mi][ni] = __builtin_amdgcn_wmma_f32_16x16x32_bf16(
                    false, al, false, bh[ni], (short)0, acc[mi][ni], false, false);
            }
        }
        __syncthreads();
    }

    // ---- epilogue (C/D layout: VGPR i = rows i / i+8, lane%16 = col) ----
#pragma unroll
    for (int mi = 0; mi < 4; ++mi)
#pragma unroll
        for (int ni = 0; ni < 2; ++ni)
#pragma unroll
            for (int i = 0; i < 8; ++i) {
                long row = bm * 128 + wm * 64 + mi * 16 + half * 8 + i;
                long col = bn * 128 + wn * 32 + ni * 16 + l15;
                float v = acc[mi][ni][i];
                if (EPI == 0) {                          // relu^2 -> bf16 hi/lo
                    v = v > 0.f ? v * v : 0.f;
                    __bf16 h = hi_bf16(v);
                    __bf16 l = lo_bf16(v, h);
                    Dh[row * N + col] = h;
                    Dl[row * N + col] = l;
                } else if (EPI == 1) {
                    Df32[row * N + col] = v;
                } else {                                 // sigmoid * kv
                    v = 1.f / (1.f + __expf(-v));
                    Df32[row * N + col] = v * Cmul[row * N + col];
                }
            }
}

// ======================================================================
// combine: scatter slot rows back to token order (0 if dropped)
// ======================================================================
__global__ __launch_bounds__(256)
void scatter_kernel(const float* __restrict__ prod,
                    const int* __restrict__ flat_idx,
                    float* __restrict__ out) {
    const int s  = blockIdx.x;
    const int fi = flat_idx[s];
    const int j  = threadIdx.x * 4;
    float4 v = (fi >= 0) ? *(const float4*)(prod + (long)fi * Mdim + j)
                         : make_float4(0.f, 0.f, 0.f, 0.f);
    *(float4*)(out + (long)s * Mdim + j) = v;
}

__global__ __launch_bounds__(256)
void shift_kernel(const float* __restrict__ x, float* __restrict__ dst) {
    const int b = blockIdx.x;
    const int j = threadIdx.x * 4;
    *(float4*)(dst + (long)b * Mdim + j) =
        *(const float4*)(x + ((long)b * Tt + (Tt - 1)) * Mdim + j);
}

// ======================================================================
extern "C" void kernel_launch(void* const* d_in, const int* in_sizes, int n_in,
                              void* d_out, int out_size, void* d_ws, size_t ws_size,
                              hipStream_t stream) {
    const float* x      = (const float*)d_in[0];
    const int*   tok    = (const int*)d_in[1];
    const float* shifts = (const float*)d_in[2];
    const float* maa_k  = (const float*)d_in[3];
    const float* maa_r  = (const float*)d_in[4];
    const float* Wk     = (const float*)d_in[5];
    const float* Wv     = (const float*)d_in[6];
    const float* Wr     = (const float*)d_in[7];
    float* out = (float*)d_out;

    // ---- workspace layout ----
    char* ws = (char*)d_ws;
    size_t off = 0;
    int* flat_idx = (int*)(ws + off);            off += (size_t)S * 4;       // 64K
    int* inv      = (int*)(ws + off);            off += (size_t)S * 4;       // 64K
    __bf16* h_hi  = (__bf16*)(ws + off);         off += (size_t)S * Fdim * 2; // 64MB
    __bf16* h_lo  = (__bf16*)(ws + off);         off += (size_t)S * Fdim * 2; // 64MB
    float*  kv    = (float*)(ws + off);          off += (size_t)S * Mdim * 4; // 64MB
    const size_t nWk = (size_t)Ee * Mdim * Fdim;
    const size_t nWv = (size_t)Ee * Fdim * Mdim;
    const size_t nWr = (size_t)Ee * Mdim * Mdim;
    __bf16* wtk_h = (__bf16*)(ws + off);         off += nWk * 2;
    __bf16* wtv_h = (__bf16*)(ws + off);         off += nWv * 2;
    __bf16* wtr_h = (__bf16*)(ws + off);         off += nWr * 2;
    __bf16* wtk_l = (__bf16*)(ws + off);         off += nWk * 2;
    __bf16* wtv_l = (__bf16*)(ws + off);         off += nWv * 2;
    __bf16* wtr_l = (__bf16*)(ws + off);         off += nWr * 2;
    float*  prod  = (float*)h_hi;                // reuse h_hi region (64MB) after GEMM2

    // 0) weight transpose + bf16 hi/lo split  (W[K][N] -> Wt[N][K])
    wsplit_kernel<<<dim3(Fdim / 32, Mdim / 32, Ee), 256, 0, stream>>>(Wk, wtk_h, wtk_l, Mdim, Fdim);
    wsplit_kernel<<<dim3(Mdim / 32, Fdim / 32, Ee), 256, 0, stream>>>(Wv, wtv_h, wtv_l, Fdim, Mdim);
    wsplit_kernel<<<dim3(Mdim / 32, Mdim / 32, Ee), 256, 0, stream>>>(Wr, wtr_h, wtr_l, Mdim, Mdim);

    // 1) routing
    route_kernel<<<1, 256, 0, stream>>>(tok, flat_idx, inv);

    // 2) h = relu^2( dispatch(xk) @ Wk )   [M=16384, K=1024, N=2048] -> bf16 hi/lo
    gemm_bf16x3<0, 0><<<dim3(Fdim / 128, S / 128), 256, 0, stream>>>(
        x, nullptr, nullptr, inv, shifts, maa_k, wtk_h, wtk_l,
        nullptr, nullptr, h_hi, h_lo, Mdim, Fdim);

    // 3) kv = h @ Wv                       [M=16384, K=2048, N=1024] -> f32
    gemm_bf16x3<1, 1><<<dim3(Mdim / 128, S / 128), 256, 0, stream>>>(
        nullptr, h_hi, h_lo, nullptr, nullptr, nullptr, wtv_h, wtv_l,
        nullptr, kv, nullptr, nullptr, Fdim, Mdim);

    // 4) prod = sigmoid( dispatch(xr) @ Wr ) * kv   [M=16384, K=1024, N=1024]
    gemm_bf16x3<0, 2><<<dim3(Mdim / 128, S / 128), 256, 0, stream>>>(
        x, nullptr, nullptr, inv, shifts, maa_r, wtr_h, wtr_l,
        kv, prod, nullptr, nullptr, Mdim, Mdim);

    // 5) combine back to token order
    scatter_kernel<<<S, 256, 0, stream>>>(prod, flat_idx, out);

    // 6) new_shift_state = x[:, -1]
    shift_kernel<<<Bb, 256, 0, stream>>>(x, out + (size_t)S * Mdim);
}
